// MolGraphModel_80109730005332
// MI455X (gfx1250) — matrix operand order repaired and checked by reading, not compiled, not run on hardware
//
#include <hip/hip_runtime.h>
#include <hip/hip_bf16.h>
#include <math.h>

// Problem dims (match reference)
constexpr int cB = 32, cN = 128, cF = 64, cE = 16, cM = 64, cD = 4, cO = 128, cT = 3;

typedef __attribute__((ext_vector_type(16))) _Float16 v16h;
typedef __attribute__((ext_vector_type(8)))  float    v8f;
typedef int v4i __attribute__((__vector_size__(16)));

#define AS1 __attribute__((address_space(1)))
#define AS3 __attribute__((address_space(3)))

// ---------------------------------------------------------------------------
// gfx1250 async global->LDS support (ASYNCcnt-tracked copy engine path)
// Prototype (from toolchain diagnostic): param0 = v4i AS1*, param1 = v4i AS3*.
// ---------------------------------------------------------------------------
#if __has_builtin(__builtin_amdgcn_global_load_async_to_lds_b128)
#define HAVE_ASYNC_LDS 1
#else
#define HAVE_ASYNC_LDS 0
#endif

#if __has_builtin(__builtin_amdgcn_s_wait_asynccnt)
#define ASYNC_WAIT() __builtin_amdgcn_s_wait_asynccnt(0)
#else
#define ASYNC_WAIT() asm volatile("s_wait_asynccnt 0x0" ::: "memory")
#endif

// ---------------------------------------------------------------------------
// WMMA helpers (CDNA5 wave32, V_WMMA_F32_16X16X32_F16)
// ---------------------------------------------------------------------------
__device__ __forceinline__ v8f wmma_f16(v16h a, v16h b, v8f c) {
  return __builtin_amdgcn_wmma_f32_16x16x32_f16(
      /*neg_a=*/false, a, /*neg_b=*/false, b,
      /*c_mod=*/(short)0, c, /*reuse_a=*/false, /*reuse_b=*/false);
}

// A fragment: 16 (rows) x 32 (K) f16.  Layout: lanes 0-15 hold K 0-15,
// lanes 16-31 hold K 16-31 (K = base + e), matching B.
__device__ __forceinline__ v16h fragA_lds(const _Float16* src, int ld) {
  int lane = threadIdx.x & 31;
  int row  = lane & 15;
  int kb   = (lane >> 4) << 4;  // 0 or 16
  v16h a;
#pragma unroll
  for (int e = 0; e < 16; ++e) {
    int k = kb + e;
    a[e] = src[row * ld + k];
  }
  return a;
}

__device__ __forceinline__ v16h fragA_g(const float* src, int ld) {
  int lane = threadIdx.x & 31;
  int row  = lane & 15;
  int kb   = (lane >> 4) << 4;
  v16h a;
#pragma unroll
  for (int e = 0; e < 16; ++e) {
    int k = kb + e;
    a[e] = (_Float16)src[(size_t)row * ld + k];
  }
  return a;
}

// B fragment: 32 (K) x 16 (cols) f16, row-major f32 source with stride ld.
// Layout used here (same as A): lanes 0-15 hold K 0-15, lanes 16-31 hold K 16-31 (K = base + e).
__device__ __forceinline__ v16h fragB_g(const float* src, int ld) {
  int lane = threadIdx.x & 31;
  int n    = lane & 15;
  int kb   = (lane >> 4) << 4;  // 0 or 16
  v16h b;
#pragma unroll
  for (int e = 0; e < 16; ++e)
    b[e] = (_Float16)src[(size_t)(kb + e) * ld + n];
  return b;
}

__device__ __forceinline__ float sigmoidf(float x) {
  return 1.0f / (1.0f + __expf(-x));
}

// ---------------------------------------------------------------------------
// Kernel 1: Hn = h @ Wn[t] + bn[t]        [B*N,F] x [F,M] -> [B*N,M]
// grid (B*N/16, M/16), block 32 (1 wave per 16x16 tile)
// ---------------------------------------------------------------------------
__global__ void k_node_pre(const float* __restrict__ h, const float* __restrict__ Wn,
                           const float* __restrict__ bn, float* __restrict__ Hn) {
  int rt = blockIdx.x, ct = blockIdx.y;
  const float* Arow = h + (size_t)rt * 16 * cF;
  const float* Bcol = Wn + ct * 16;

  v8f acc = {};
  acc = wmma_f16(fragA_g(Arow, cF),            fragB_g(Bcol, cM),            acc);
  acc = wmma_f16(fragA_g(Arow + 32, cF),       fragB_g(Bcol + 32 * cM, cM),  acc);

  int lane = threadIdx.x & 31;
  int n  = lane & 15;
  int mb = (lane >> 4) << 3;
#pragma unroll
  for (int v = 0; v < 8; ++v) {
    int row = rt * 16 + mb + v;
    int col = ct * 16 + n;
    Hn[(size_t)row * cM + col] = acc[v] + bn[col];
  }
}

// ---------------------------------------------------------------------------
// Kernel 2: ai = h@wai, aj = h@waj  (per node)
// ---------------------------------------------------------------------------
__global__ void k_attn_scalars(const float* __restrict__ h, const float* __restrict__ wai,
                               const float* __restrict__ waj, float* __restrict__ ai,
                               float* __restrict__ aj) {
  int r = blockIdx.x * blockDim.x + threadIdx.x;  // 0 .. B*N-1
  float s1 = 0.f, s2 = 0.f;
#pragma unroll
  for (int f = 0; f < cF; ++f) {
    float hv = h[(size_t)r * cF + f];
    s1 += hv * wai[f];
    s2 += hv * waj[f];
  }
  ai[r] = s1;
  aj[r] = s2;
}

// ---------------------------------------------------------------------------
// Kernel 3: fused edge gate + aggregation.
//   w_ij = (sum_d alpha_d adj[b,d,i,j]) * sigmoid(ai_i + aj_j + bfm_ij.wae + ba)
//   agg[b,i,:] = (sum_j w_ij bfm_ij) @ We + (sum_j w_ij) * be + W_row @ Hn[b]
// Hn[b] (32 KB) is staged into LDS with the gfx1250 async copy path while the
// waves build the gate matrix (overlaps ASYNC DMA with VALU transcendentals),
// then WMMA consumes both operands from LDS.
// grid (B, N/16), block 128 (4 waves)
// ---------------------------------------------------------------------------
__global__ void k_edge_agg(const float* __restrict__ Hn, const float* __restrict__ bfm,
                           const float* __restrict__ adj, const float* __restrict__ ai,
                           const float* __restrict__ aj, const float* __restrict__ wae,
                           const float* __restrict__ ba_p, const float* __restrict__ aggw,
                           const float* __restrict__ We, const float* __restrict__ be,
                           float* __restrict__ agg) {
  __shared__ _Float16 Wlds[16][132];     // gate matrix tile, f16 (A operand)
  __shared__ float    HnS[cN * cM];      // Hn[b] staged (B operand source), 32 KB
  __shared__ float    sumw_s[16];
  __shared__ float    ebar_s[16][16];
  __shared__ float    wae_s[16];
  __shared__ float    alpha_s[cD];

  int b   = blockIdx.x;
  int it  = blockIdx.y;
  int tid = threadIdx.x;

  // Kick off async staging of Hn[b] -> LDS (2048 float4s, 16 per thread)
  const float* HnB = Hn + (size_t)b * cN * cM;
#if HAVE_ASYNC_LDS
#pragma unroll
  for (int q = 0; q < 16; ++q) {
    int idx = q * 128 + tid;  // one 16B packet each
    __builtin_amdgcn_global_load_async_to_lds_b128(
        (AS1 v4i*)(HnB + idx * 4),
        (AS3 v4i*)(HnS + idx * 4), 0, 0);
  }
#else
  for (int idx = tid; idx < cN * cM; idx += 128) HnS[idx] = HnB[idx];
#endif

  if (tid < 16) { wae_s[tid] = wae[tid]; sumw_s[tid] = 0.f; }
  if (tid == 0) {
    float mx = aggw[0];
    for (int d = 1; d < cD; ++d) mx = fmaxf(mx, aggw[d]);
    float ex[cD], s = 0.f;
    for (int d = 0; d < cD; ++d) { ex[d] = __expf(aggw[d] - mx); s += ex[d]; }
    for (int d = 0; d < cD; ++d) alpha_s[d] = ex[d] / s;
  }
  ((float*)ebar_s)[tid] = 0.f;
  ((float*)ebar_s)[tid + 128] = 0.f;
  __syncthreads();

  // Phase 1: build gate row tile [16 x 128] + edge reductions (overlaps DMA)
  int il = tid >> 3;         // local i: 0..15
  int jg = tid & 7;          // j group of 16
  int i  = it * 16 + il;
  float a_i  = ai[b * cN + i];
  float ba_v = ba_p[0];

  float lsum = 0.f;
  float leb[cE];
#pragma unroll
  for (int e = 0; e < cE; ++e) leb[e] = 0.f;

  for (int jj = 0; jj < 16; ++jj) {
    int j = jg * 16 + jj;
    const float* bptr = bfm + (((size_t)(b * cN + i)) * cN + j) * cE;
    __builtin_prefetch(bptr + cE, 0, 3);   // next edge row of the bfm stream
    float ev[cE];
    float sc = a_i + aj[b * cN + j] + ba_v;
#pragma unroll
    for (int e = 0; e < cE; ++e) { ev[e] = bptr[e]; sc += ev[e] * wae_s[e]; }
    float A = 0.f;
#pragma unroll
    for (int d = 0; d < cD; ++d)
      A += alpha_s[d] * adj[(((size_t)(b * cD + d)) * cN + i) * cN + j];
    float w = A * sigmoidf(sc);
    Wlds[il][j] = (_Float16)w;
    lsum += w;
#pragma unroll
    for (int e = 0; e < cE; ++e) leb[e] += w * ev[e];
  }
  atomicAdd(&sumw_s[il], lsum);
#pragma unroll
  for (int e = 0; e < cE; ++e) atomicAdd(&ebar_s[il][e], leb[e]);

#if HAVE_ASYNC_LDS
  ASYNC_WAIT();  // this wave's async packets landed in LDS
#endif
  __syncthreads();

  // Phase 2: aggH tile = Wlds[16x128] @ HnS[128 x 16cols], one wave per m-tile
  int wave = tid >> 5;   // 0..3 -> which 16 columns of M
  v8f acc = {};
#pragma unroll
  for (int kc = 0; kc < 4; ++kc) {
    v16h a = fragA_lds(&Wlds[0][kc * 32], 132);
    v16h bfr;
    {
      int lane = tid & 31;
      int n    = lane & 15;
      int kb   = (lane >> 4) << 4;
      const float* src = HnS + (size_t)(kc * 32) * cM + wave * 16;
#pragma unroll
      for (int e = 0; e < 16; ++e)
        bfr[e] = (_Float16)src[(kb + e) * cM + n];
    }
    acc = wmma_f16(a, bfr, acc);
  }

  int lane = tid & 31;
  int n  = lane & 15;
  int mb = (lane >> 4) << 3;
#pragma unroll
  for (int v = 0; v < 8; ++v) {
    int irow = mb + v;                 // local i
    int mcol = wave * 16 + n;          // 0..63
    float s = acc[v] + sumw_s[irow] * be[mcol];
#pragma unroll
    for (int e = 0; e < cE; ++e) s += ebar_s[irow][e] * We[e * cM + mcol];
    agg[((size_t)(b * cN) + it * 16 + irow) * cM + mcol] = s;
  }
}

// ---------------------------------------------------------------------------
// Kernel 4: GRU update, fully fused per 16-node tile.
//   zr = sigmoid(agg@gW[:, :2F] + h@gU[:, :2F] + gb[:2F])
//   n  = tanh  (agg@gW[:,2F:] + (r*h)@gU[:,2F:] + gb[2F:])
//   hout = mask * ((1-z)*n + z*h)
// grid B*N/16, block 128 (4 waves)
// ---------------------------------------------------------------------------
__global__ void k_gru(const float* __restrict__ agg, const float* __restrict__ hin,
                      const float* __restrict__ mask, const float* __restrict__ gW,
                      const float* __restrict__ gU, const float* __restrict__ gb,
                      float* __restrict__ hout) {
  __shared__ _Float16 aggA[16][72];
  __shared__ _Float16 hA16[16][72];
  __shared__ float    hF[16][64];
  __shared__ float    zr[16][128];
  __shared__ _Float16 rh16[16][72];
  __shared__ float    msk[16];

  int rt  = blockIdx.x;
  int tid = threadIdx.x;

  for (int idx = tid; idx < 16 * 64; idx += 128) {
    int i = idx >> 6, f = idx & 63;
    size_t row = (size_t)(rt * 16 + i);
    float av = agg[row * cM + f];
    float hv = hin[row * cF + f];
    aggA[i][f] = (_Float16)av;
    hA16[i][f] = (_Float16)hv;
    hF[i][f]   = hv;
  }
  if (tid < 16) msk[tid] = mask[rt * 16 + tid];
  __syncthreads();

  int wave = tid >> 5, lane = tid & 31;
  int n  = lane & 15;
  int mb = (lane >> 4) << 3;

  // Phase 1: z|r columns 0..127, two 16-col tiles per wave
#pragma unroll
  for (int q = 0; q < 2; ++q) {
    int colb = (wave * 2 + q) * 16;
    float gbv = gb[colb + n];
    v8f acc;
#pragma unroll
    for (int v = 0; v < 8; ++v) acc[v] = gbv;
#pragma unroll
    for (int kc = 0; kc < 2; ++kc) {
      acc = wmma_f16(fragA_lds(&aggA[0][kc * 32], 72),
                     fragB_g(gW + (size_t)kc * 32 * (3 * cF) + colb, 3 * cF), acc);
      acc = wmma_f16(fragA_lds(&hA16[0][kc * 32], 72),
                     fragB_g(gU + (size_t)kc * 32 * (3 * cF) + colb, 3 * cF), acc);
    }
#pragma unroll
    for (int v = 0; v < 8; ++v)
      zr[mb + v][colb + n] = sigmoidf(acc[v]);
  }
  __syncthreads();

  // Phase 2: rh = r * h (f16 operand for candidate GEMM)
  for (int idx = tid; idx < 16 * 64; idx += 128) {
    int i = idx >> 6, f = idx & 63;
    rh16[i][f] = (_Float16)(zr[i][64 + f] * hF[i][f]);
  }
  __syncthreads();

  // Phase 3: candidate, columns 128..191, one 16-col tile per wave
  {
    int colb = 128 + wave * 16;
    float gbv = gb[colb + n];
    v8f acc;
#pragma unroll
    for (int v = 0; v < 8; ++v) acc[v] = gbv;
#pragma unroll
    for (int kc = 0; kc < 2; ++kc) {
      acc = wmma_f16(fragA_lds(&aggA[0][kc * 32], 72),
                     fragB_g(gW + (size_t)kc * 32 * (3 * cF) + colb, 3 * cF), acc);
      acc = wmma_f16(fragA_lds(&rh16[0][kc * 32], 72),
                     fragB_g(gU + (size_t)kc * 32 * (3 * cF) + colb, 3 * cF), acc);
    }
#pragma unroll
    for (int v = 0; v < 8; ++v) {
      int i = mb + v;
      int f = wave * 16 + n;
      float nv = tanhf(acc[v]);
      float z  = zr[i][f];
      float hv = hF[i][f];
      hout[(size_t)(rt * 16 + i) * cF + f] = msk[i] * ((1.f - z) * nv + z * hv);
    }
  }
}

// ---------------------------------------------------------------------------
// Kernel 5: readout energies e[b,n] = tanh(h@roWa + roba) @ rov (masked)
// ---------------------------------------------------------------------------
__global__ void k_readout_e(const float* __restrict__ h, const float* __restrict__ mask,
                            const float* __restrict__ roWa, const float* __restrict__ roba,
                            const float* __restrict__ rov, float* __restrict__ esc) {
  int r = blockIdx.x * blockDim.x + threadIdx.x;  // 0 .. B*N-1
  float hr[cF];
#pragma unroll
  for (int f = 0; f < cF; ++f) hr[f] = h[(size_t)r * cF + f];
  float acc = 0.f;
  for (int fp = 0; fp < cF; ++fp) {
    float s = roba[fp];
#pragma unroll
    for (int f = 0; f < cF; ++f) s += hr[f] * roWa[f * cF + fp];
    acc += tanhf(s) * rov[fp];
  }
  esc[r] = (mask[r] > 0.f) ? acc : -1e9f;
}

// ---------------------------------------------------------------------------
// Kernel 6: per-batch softmax over nodes + attention pooling
// grid B, block 128 (== N)
// ---------------------------------------------------------------------------
__global__ void k_softmax_pool(const float* __restrict__ esc, const float* __restrict__ h,
                               float* __restrict__ pooled) {
  __shared__ float sh[128];
  __shared__ float att[128];
  int b = blockIdx.x, tid = threadIdx.x;
  float v = esc[b * cN + tid];
  sh[tid] = v;
  __syncthreads();
  for (int s = 64; s > 0; s >>= 1) {
    if (tid < s) sh[tid] = fmaxf(sh[tid], sh[tid + s]);
    __syncthreads();
  }
  float mx = sh[0];
  __syncthreads();
  float ex = __expf(v - mx);
  sh[tid] = ex;
  __syncthreads();
  for (int s = 64; s > 0; s >>= 1) {
    if (tid < s) sh[tid] += sh[tid + s];
    __syncthreads();
  }
  att[tid] = ex / sh[0];
  __syncthreads();
  if (tid < cF) {
    float s = 0.f;
#pragma unroll 8
    for (int n = 0; n < cN; ++n) s += att[n] * h[((size_t)(b * cN) + n) * cF + tid];
    pooled[b * cF + tid] = s;
  }
}

// ---------------------------------------------------------------------------
// Kernel 7: out = pooled @ roWo + robo   [B,F]x[F,O]
// ---------------------------------------------------------------------------
__global__ void k_out(const float* __restrict__ pooled, const float* __restrict__ roWo,
                      const float* __restrict__ robo, float* __restrict__ out) {
  int idx = blockIdx.x * blockDim.x + threadIdx.x;  // 0 .. B*O-1
  int b = idx >> 7, o = idx & 127;
  float s = robo[o];
#pragma unroll
  for (int f = 0; f < cF; ++f) s += pooled[b * cF + f] * roWo[f * cO + o];
  out[idx] = s;
}

// ---------------------------------------------------------------------------
// Host launcher
// ---------------------------------------------------------------------------
extern "C" void kernel_launch(void* const* d_in, const int* in_sizes, int n_in,
                              void* d_out, int out_size, void* d_ws, size_t ws_size,
                              hipStream_t stream) {
  (void)in_sizes; (void)n_in; (void)out_size; (void)ws_size;

  const float* afm  = (const float*)d_in[0];
  const float* bfm  = (const float*)d_in[1];
  const float* mask = (const float*)d_in[2];
  const float* adj  = (const float*)d_in[3];
  const float* We   = (const float*)d_in[4];
  const float* be   = (const float*)d_in[5];
  const float* Wn   = (const float*)d_in[6];
  const float* bn   = (const float*)d_in[7];
  const float* wai  = (const float*)d_in[8];
  const float* waj  = (const float*)d_in[9];
  const float* wae  = (const float*)d_in[10];
  const float* ba   = (const float*)d_in[11];
  const float* aggw = (const float*)d_in[12];
  const float* gW   = (const float*)d_in[13];
  const float* gU   = (const float*)d_in[14];
  const float* gb   = (const float*)d_in[15];
  const float* roWa = (const float*)d_in[16];
  const float* roba = (const float*)d_in[17];
  const float* rov  = (const float*)d_in[18];
  const float* roWo = (const float*)d_in[19];
  const float* robo = (const float*)d_in[20];
  float* out = (float*)d_out;

  float* ws     = (float*)d_ws;
  float* hBufA  = ws;                      // [B*N*F]
  float* hBufB  = hBufA + cB * cN * cF;    // [B*N*F]
  float* Hn     = hBufB + cB * cN * cF;    // [B*N*M]
  float* ai     = Hn + cB * cN * cM;       // [B*N]
  float* aj     = ai + cB * cN;            // [B*N]
  float* agg    = aj + cB * cN;            // [B*N*M]
  float* esc    = agg + cB * cN * cM;      // [B*N]
  float* pooled = esc + cB * cN;           // [B*F]

  const float* hcur = afm;
  float* bufs[2] = { hBufA, hBufB };

  for (int t = 0; t < cT; ++t) {
    const float* Wn_t   = Wn   + (size_t)t * cF * cM;
    const float* bn_t   = bn   + (size_t)t * cM;
    const float* wai_t  = wai  + (size_t)t * cF;
    const float* waj_t  = waj  + (size_t)t * cF;
    const float* wae_t  = wae  + (size_t)t * cE;
    const float* ba_t   = ba   + t;
    const float* aggw_t = aggw + (size_t)t * cD;
    const float* We_t   = We   + (size_t)t * cE * cM;
    const float* be_t   = be   + (size_t)t * cM;
    const float* gW_t   = gW   + (size_t)t * cM * 3 * cF;
    const float* gU_t   = gU   + (size_t)t * cF * 3 * cF;
    const float* gb_t   = gb   + (size_t)t * 3 * cF;
    float* hnext = bufs[t & 1];   // t0->A, t1->B, t2->A

    k_node_pre<<<dim3(cB * cN / 16, cM / 16), 32, 0, stream>>>(hcur, Wn_t, bn_t, Hn);
    k_attn_scalars<<<cB * cN / 128, 128, 0, stream>>>(hcur, wai_t, waj_t, ai, aj);
    k_edge_agg<<<dim3(cB, cN / 16), 128, 0, stream>>>(Hn, bfm, adj, ai, aj, wae_t,
                                                      ba_t, aggw_t, We_t, be_t, agg);
    k_gru<<<cB * cN / 16, 128, 0, stream>>>(agg, hcur, mask, gW_t, gU_t, gb_t, hnext);
    hcur = hnext;
  }

  k_readout_e<<<cB * cN / 128, 128, 0, stream>>>(hcur, mask, roWa, roba, rov, esc);
  k_softmax_pool<<<cB, 128, 0, stream>>>(esc, hcur, pooled);
  k_out<<<cB * cO / 128, 128, 0, stream>>>(pooled, roWo, robo, out);
}
